// GraphAttentionLayer_15272903704636
// MI455X (gfx1250) — matrix operand order repaired and checked by reading, not compiled, not run on hardware
//
#include <hip/hip_runtime.h>
#include <hip/hip_bf16.h>
#include <math.h>

typedef __attribute__((ext_vector_type(16))) __bf16       v16bf;
typedef __attribute__((ext_vector_type(8)))  float        v8f;
typedef __attribute__((ext_vector_type(4)))  unsigned int u32x4;
typedef __attribute__((ext_vector_type(8)))  int          i32x8;
typedef __attribute__((ext_vector_type(4)))  int          i32x4;

#define D_FEAT    128
#define NUM_HEADS 4
#define HEAD_DIM  32
#define MAX_K     512   // max GEMM K staged in LDS (16 x 528 dwords incl. padding)

__device__ __forceinline__ void atomicMaxFloat(float* addr, float val) {
    // works for any sign when addr is initialized to -inf
    if (val >= 0.0f) atomicMax((int*)addr, __float_as_int(val));
    else             atomicMin((unsigned int*)addr, __float_as_uint(val));
}

// ---------------------------------------------------------------------------
// Weight pre-pack: fp32 B[K,N] -> bf16 fragment-major layout.
// One thread per (kTile,nTile,lane): writes the 16 bf16 values (32 contiguous
// bytes) that lane reads for v_wmma_f32_16x16x32_bf16:
//   b[j] = B[kTile*32 + (lane>>4)*16 + j][nTile*16 + (lane&15)]
// ---------------------------------------------------------------------------
__global__ void pack_b_kernel(const float* __restrict__ B, __bf16* __restrict__ Bp,
                              int K, int N)
{
    const int t  = blockIdx.x * blockDim.x + threadIdx.x;
    const int nT = N >> 4;
    const int total = (K >> 5) * nT * 32;
    if (t >= total) return;
    const int lane = t & 31;
    const int tile = t >> 5;
    const int nt   = tile % nT;
    const int kt   = tile / nT;
    const int col  = nt * 16 + (lane & 15);
    const int kb   = kt * 32 + (lane >> 4) * 16;
    __bf16* dst = Bp + (size_t)t * 16;
#pragma unroll
    for (int j = 0; j < 16; ++j)
        dst[j] = (__bf16)B[(size_t)(kb + j) * N + col];
}

// ---------------------------------------------------------------------------
// WMMA GEMM: C[M,N] = A[M,K] @ B[K,N] + bias.
//  * A (fp32, row-major) staged per block via the Tensor Data Mover into LDS
//    (16 x K stripe, TDM row padding: +4 dwords per 128 dwords -> conflict-free
//    column reads on 64 banks), then read with ds_load_b128 and cvt to bf16.
//  * B pre-packed bf16 fragments: 2x global_load_b128 per wave per K-step.
//  * epilogue: 0 = bias; 1 = bias + gelu(tanh); 2 = bias + residual add.
//  Block = 256 threads = 8 waves; wave w owns 16x16 tile (blockIdx.x, y*8+w).
//  Requires K % 32 == 0, N % 16 == 0, K <= MAX_K.
// ---------------------------------------------------------------------------
__global__ void __launch_bounds__(256)
wmma_gemm_kernel(const float* __restrict__ A, const __bf16* __restrict__ Bp,
                 const float* __restrict__ bias, const float* __restrict__ resid,
                 float* __restrict__ C, int M, int N, int K, int epilogue)
{
    __shared__ float atile[16 * (MAX_K + (MAX_K >> 7) * 4)];  // 16 x 528 dwords

    const int wave  = threadIdx.x >> 5;
    const int lane  = threadIdx.x & 31;
    const int mTile = blockIdx.x;
    const int nTile = blockIdx.y * 8 + wave;
    const int rowStride = K + ((K >> 7) << 2);   // padded LDS row stride (dwords)

    // ---- TDM: DMA the 16 x K fp32 A-stripe into LDS (wave 0 issues) ----
    if (threadIdx.x < 32) {
        const uint64_t gaddr  = (uint64_t)(uintptr_t)(A + (size_t)mTile * 16 * K);
        const uint32_t ldsoff = (uint32_t)(uintptr_t)&atile[0];  // low 32b = LDS offset
        u32x4 g0;
        g0.x = 1u;                                    // count=1, user descriptor
        g0.y = ldsoff;                                // lds_addr
        g0.z = (uint32_t)gaddr;                       // global_addr[31:0]
        g0.w = ((uint32_t)(gaddr >> 32) & 0x01FFFFFFu) | (2u << 30);  // addr[56:32] | type=2
        const uint32_t td0 = (uint32_t)K;             // tensor_dim0 (row length, elems)
        const uint32_t td1 = (uint32_t)M;             // tensor_dim1 (rows; OOB reads 0)
        const uint64_t s0  = (uint64_t)K;             // tensor_dim0_stride (elems)
        const uint64_t s1  = (uint64_t)K * 16u;       // tensor_dim1_stride (unused, 2D)
        i32x8 g1;
        g1[0] = (int)((2u << 16)      // data_size = 4B
                    | (1u << 20)      // pad_enable
                    | (6u << 22)      // pad_interval: every 128 dwords
                    | (3u << 25));    // pad_amount: 4 dwords
        g1[1] = (int)((td0 & 0xFFFFu) << 16);                       // [63:48]=td0.lo
        g1[2] = (int)((td0 >> 16) | ((td1 & 0xFFFFu) << 16));       // td0.hi | td1.lo
        g1[3] = (int)((td1 >> 16) | ((uint32_t)K << 16));           // td1.hi | tile_dim0=K
        g1[4] = 16;                                                 // tile_dim1=16, tile_dim2=0
        g1[5] = (int)(uint32_t)s0;                                  // stride0[31:0]
        g1[6] = (int)(((uint32_t)(s0 >> 32) & 0xFFFFu)
                    | (((uint32_t)s1 & 0xFFFFu) << 16));            // stride0.hi | stride1.lo
        g1[7] = (int)(uint32_t)(s1 >> 16);                          // stride1[47:16]
        i32x4 gz4 = {0, 0, 0, 0};
        i32x8 gz8 = {0, 0, 0, 0, 0, 0, 0, 0};
        __builtin_amdgcn_tensor_load_to_lds(g0, g1, gz4, gz4, gz8, 0);
        __builtin_amdgcn_s_wait_tensorcnt(0);
    }
    __syncthreads();

    if (nTile * 16 >= N) return;

    const int hi    = lane >> 4;                 // half-wave select
    const int arow  = (lane & 15) * rowStride;   // this lane's A row in LDS
    const int col   = nTile * 16 + (lane & 15);
    const int nT    = N >> 4;

    v8f acc = {};
    for (int kk = 0; kk < K; kk += 32) {
        // A fragment: lanes 0-15 hold K=kk+[0..15]; lanes 16-31: kk+[16..31].
        float av[16];
        const int c0 = kk + hi * 16;
        const int i0 = arow + c0 + ((c0 >> 7) << 2);
        *(float4*)&av[0] = *(const float4*)&atile[i0];
        *(float4*)&av[4] = *(const float4*)&atile[i0 + 4];
        const int c1 = c0 + 8;
        const int i1 = arow + c1 + ((c1 >> 7) << 2);
        *(float4*)&av[8]  = *(const float4*)&atile[i1];
        *(float4*)&av[12] = *(const float4*)&atile[i1 + 4];
        v16bf a;
#pragma unroll
        for (int i = 0; i < 16; ++i) a[i] = (__bf16)av[i];

        // B fragment: 32 contiguous bytes per lane from the packed layout.
        const v16bf b = *(const v16bf*)(Bp +
            (((size_t)(kk >> 5) * nT + nTile) * 32 + lane) * 16);

        acc = __builtin_amdgcn_wmma_f32_16x16x32_bf16(
                  false, a, false, b, (short)0, acc, false, false);
    }

    // C/D layout: VGPR r -> M = mTile*16 + hi*8 + r, N = col
    const float bvv   = bias[col];
    const int   mBase = mTile * 16 + hi * 8;
#pragma unroll
    for (int r = 0; r < 8; ++r) {
        const int m = mBase + r;
        if (m >= M) break;
        float x = acc[r] + bvv;
        if (epilogue == 1) {                     // tanh-approx GELU (jax default)
            const float x3 = x * x * x;
            x = 0.5f * x * (1.0f + tanhf(0.7978845608028654f * (x + 0.044715f * x3)));
        } else if (epilogue == 2) {
            x += resid[(size_t)m * N + col];
        }
        C[(size_t)m * N + col] = x;
    }
}

// ---------------------------------------------------------------------------
// Edge phase
// ---------------------------------------------------------------------------
__global__ void init_kernel(float* __restrict__ segmax, float* __restrict__ segsum,
                            float* __restrict__ agg, int N)
{
    const int i = blockIdx.x * blockDim.x + threadIdx.x;
    if (i < N * NUM_HEADS) { segmax[i] = -__builtin_inff(); segsum[i] = 0.0f; }
    if (i < N * D_FEAT)    agg[i] = 0.0f;
}

__global__ void edge_scores_kernel(const float* __restrict__ q, const float* __restrict__ k,
                                   const int* __restrict__ ei, float* __restrict__ scores,
                                   float* __restrict__ segmax, int E)
{
    const int e = blockIdx.x * blockDim.x + threadIdx.x;
    if (e >= E) return;
    const int src = ei[e];
    const int dst = ei[E + e];
    const float4* qd = (const float4*)(q + (size_t)dst * D_FEAT);
    const float4* ks = (const float4*)(k + (size_t)src * D_FEAT);
    float acc[NUM_HEADS] = {0.f, 0.f, 0.f, 0.f};
#pragma unroll
    for (int i = 0; i < 32; ++i) {
        const float4 a = qd[i], b = ks[i];
        acc[i >> 3] += a.x * b.x + a.y * b.y + a.z * b.z + a.w * b.w;
    }
    const float scale = 0.17677669529663687f;    // 1/sqrt(32)
#pragma unroll
    for (int h = 0; h < NUM_HEADS; ++h) {
        const float s = acc[h] * scale;
        scores[(size_t)e * NUM_HEADS + h] = s;
        atomicMaxFloat(&segmax[dst * NUM_HEADS + h], s);
    }
}

__global__ void edge_exp_kernel(const int* __restrict__ ei, float* __restrict__ scores,
                                const float* __restrict__ segmax, float* __restrict__ segsum,
                                int E)
{
    const int e = blockIdx.x * blockDim.x + threadIdx.x;
    if (e >= E) return;
    const int dst = ei[E + e];
#pragma unroll
    for (int h = 0; h < NUM_HEADS; ++h) {
        const size_t idx = (size_t)e * NUM_HEADS + h;
        const float ex = expf(scores[idx] - segmax[dst * NUM_HEADS + h]);
        scores[idx] = ex;                         // reuse buffer for exp values
        atomicAdd(&segsum[dst * NUM_HEADS + h], ex);
    }
}

__global__ void edge_agg_kernel(const int* __restrict__ ei, const float* __restrict__ ex,
                                const float* __restrict__ segsum, const float* __restrict__ v,
                                float* __restrict__ agg, int E)
{
    const int t = blockIdx.x * blockDim.x + threadIdx.x;
    if (t >= E * NUM_HEADS) return;
    const int e = t >> 2, h = t & 3;
    const int src = ei[e];
    const int dst = ei[E + e];
    const float w = ex[(size_t)e * NUM_HEADS + h] / segsum[dst * NUM_HEADS + h];
    const float* vp = v   + (size_t)src * D_FEAT + h * HEAD_DIM;
    float*       ap = agg + (size_t)dst * D_FEAT + h * HEAD_DIM;
#pragma unroll
    for (int d = 0; d < HEAD_DIM; ++d)
        atomicAdd(ap + d, w * vp[d]);
}

// ---------------------------------------------------------------------------
// LayerNorm(out + residual): one wave32 per row of 128 elements
// ---------------------------------------------------------------------------
__global__ void __launch_bounds__(256)
ln_kernel(const float* __restrict__ proj, const float* __restrict__ resid,
          const float* __restrict__ g, const float* __restrict__ b,
          float* __restrict__ out, int N)
{
    const int wave = threadIdx.x >> 5;
    const int lane = threadIdx.x & 31;
    const int row  = blockIdx.x * 8 + wave;
    if (row >= N) return;
    const float* p = proj  + (size_t)row * D_FEAT;
    const float* r = resid + (size_t)row * D_FEAT;
    float x[4];
    float s = 0.0f;
#pragma unroll
    for (int i = 0; i < 4; ++i) {
        const int c = lane + 32 * i;
        x[i] = p[c] + r[c];
        s += x[i];
    }
#pragma unroll
    for (int off = 16; off > 0; off >>= 1) s += __shfl_xor(s, off, 32);
    const float mu = s * (1.0f / 128.0f);
    float vv = 0.0f;
#pragma unroll
    for (int i = 0; i < 4; ++i) { const float d = x[i] - mu; vv += d * d; }
#pragma unroll
    for (int off = 16; off > 0; off >>= 1) vv += __shfl_xor(vv, off, 32);
    const float rs = rsqrtf(vv * (1.0f / 128.0f) + 1e-5f);
#pragma unroll
    for (int i = 0; i < 4; ++i) {
        const int c = lane + 32 * i;
        out[(size_t)row * D_FEAT + c] = (x[i] - mu) * rs * g[c] + b[c];
    }
}

// ---------------------------------------------------------------------------
extern "C" void kernel_launch(void* const* d_in, const int* in_sizes, int n_in,
                              void* d_out, int out_size, void* d_ws, size_t ws_size,
                              hipStream_t stream)
{
    const float* X   = (const float*)d_in[0];
    const int*   ei  = (const int*)  d_in[1];
    const float* Wq  = (const float*)d_in[2];
    const float* bq  = (const float*)d_in[3];
    const float* Wk  = (const float*)d_in[4];
    const float* bk  = (const float*)d_in[5];
    const float* Wv  = (const float*)d_in[6];
    const float* bv  = (const float*)d_in[7];
    const float* Wo  = (const float*)d_in[8];
    const float* bo  = (const float*)d_in[9];
    const float* lng = (const float*)d_in[10];
    const float* lnb = (const float*)d_in[11];
    const float* W1  = (const float*)d_in[12];
    const float* b1  = (const float*)d_in[13];
    const float* W2  = (const float*)d_in[14];
    const float* b2  = (const float*)d_in[15];
    float* out = (float*)d_out;

    const int N = in_sizes[0] / D_FEAT;   // 40000
    const int E = in_sizes[1] / 2;        // 640000
    const int H = in_sizes[13];           // 512 (b1 length)

    // workspace layout (fp32)
    float* ws     = (float*)d_ws;
    float* q      = ws;
    float* k      = q      + (size_t)N * D_FEAT;
    float* v      = k      + (size_t)N * D_FEAT;
    float* scores = v      + (size_t)N * D_FEAT;          // E*4, reused as exp()
    float* segmax = scores + (size_t)E * NUM_HEADS;
    float* segsum = segmax + (size_t)N * NUM_HEADS;
    float* agg    = segsum + (size_t)N * NUM_HEADS;
    float* proj   = agg    + (size_t)N * D_FEAT;
    float* outln  = proj   + (size_t)N * D_FEAT;
    float* hid    = ws;    // reuse q/k/v/scores region (dead after attention)

    // packed bf16 weight buffers (32B aligned), after outln
    uintptr_t pb = (uintptr_t)(outln + (size_t)N * D_FEAT);
    pb = (pb + 31) & ~(uintptr_t)31;
    __bf16* pWq = (__bf16*)pb;
    __bf16* pWk = pWq + 128 * 128;
    __bf16* pWv = pWk + 128 * 128;
    __bf16* pWo = pWv + 128 * 128;
    __bf16* pW1 = pWo + 128 * 128;
    __bf16* pW2 = pW1 + (size_t)128 * 512;

    const dim3 blk(256);
    const int mTiles = (N + 15) / 16;
    const dim3 gD(mTiles, (D_FEAT + 127) / 128);
    const dim3 gH(mTiles, (H + 127) / 128);

    // pre-pack weights into WMMA fragment layout (bf16)
    const int tD = (D_FEAT / 32) * (D_FEAT / 16) * 32;   // 1024
    const int t1 = (D_FEAT / 32) * (H / 16) * 32;        // 4096
    const int t2 = (H / 32) * (D_FEAT / 16) * 32;        // 4096
    pack_b_kernel<<<(tD + 255) / 256, blk, 0, stream>>>(Wq, pWq, D_FEAT, D_FEAT);
    pack_b_kernel<<<(tD + 255) / 256, blk, 0, stream>>>(Wk, pWk, D_FEAT, D_FEAT);
    pack_b_kernel<<<(tD + 255) / 256, blk, 0, stream>>>(Wv, pWv, D_FEAT, D_FEAT);
    pack_b_kernel<<<(tD + 255) / 256, blk, 0, stream>>>(Wo, pWo, D_FEAT, D_FEAT);
    pack_b_kernel<<<(t1 + 255) / 256, blk, 0, stream>>>(W1, pW1, D_FEAT, H);
    pack_b_kernel<<<(t2 + 255) / 256, blk, 0, stream>>>(W2, pW2, H, D_FEAT);

    // Q/K/V projections (TDM-staged A + WMMA bf16)
    wmma_gemm_kernel<<<gD, blk, 0, stream>>>(X, pWq, bq, nullptr, q, N, D_FEAT, D_FEAT, 0);
    wmma_gemm_kernel<<<gD, blk, 0, stream>>>(X, pWk, bk, nullptr, k, N, D_FEAT, D_FEAT, 0);
    wmma_gemm_kernel<<<gD, blk, 0, stream>>>(X, pWv, bv, nullptr, v, N, D_FEAT, D_FEAT, 0);

    // edge attention (segment softmax + scatter-add), L2-resident gathers
    init_kernel<<<(N * D_FEAT + 255) / 256, blk, 0, stream>>>(segmax, segsum, agg, N);
    edge_scores_kernel<<<(E + 255) / 256, blk, 0, stream>>>(q, k, ei, scores, segmax, E);
    edge_exp_kernel   <<<(E + 255) / 256, blk, 0, stream>>>(ei, scores, segmax, segsum, E);
    edge_agg_kernel   <<<(E * NUM_HEADS + 255) / 256, blk, 0, stream>>>(ei, scores, segsum, v, agg, E);

    // output projection + residual LayerNorm
    wmma_gemm_kernel<<<gD, blk, 0, stream>>>(agg, pWo, bo, nullptr, proj, N, D_FEAT, D_FEAT, 0);
    ln_kernel<<<(N + 7) / 8, blk, 0, stream>>>(proj, X, lng, lnb, outln, N);

    // MLP: gelu(outln @ W1 + b1) @ W2 + b2 + outln
    wmma_gemm_kernel<<<gH, blk, 0, stream>>>(outln, pW1, b1, nullptr, hid, N, H, D_FEAT, 1);
    wmma_gemm_kernel<<<gD, blk, 0, stream>>>(hid, pW2, b2, outln, out, N, D_FEAT, H, 2);
}
